// ElectraSimpleAttentionOutput_3195455668298
// MI455X (gfx1250) — matrix operand order repaired and checked by reading, not compiled, not run on hardware
//
#include <hip/hip_runtime.h>
#include <hip/hip_bf16.h>

typedef __attribute__((ext_vector_type(16))) __bf16 v16bf;
typedef __attribute__((ext_vector_type(8)))  float  v8f;

#define T_TOK 65536
#define HDIM  1024
#define B_SEG 64
#define L_OUT 2
#define LN_EPS 1e-12f

#define MB     128   // rows per block
#define MG     32    // rows per m-group (2 WMMA m-tiles)
#define NWAVE  8
#define NW     128   // cols per wave
#define SEGCAP 8     // LDS segment-accumulator capacity

__device__ __forceinline__ void atomic_add_f32_global(float* p, float v) {
    __hip_atomic_fetch_add(p, v, __ATOMIC_RELAXED, __HIP_MEMORY_SCOPE_AGENT);
}
__device__ __forceinline__ void atomic_add_f32_lds(float* p, float v) {
    __hip_atomic_fetch_add(p, v, __ATOMIC_RELAXED, __HIP_MEMORY_SCOPE_WORKGROUP);
}

// ---------------------------------------------------------------------------
// One-time W pre-swizzle: fp32 [H,H] row-major -> bf16 B-fragment order
//   Wb[(((kb*64 + ntile)*32 + lane)*16 + e)] = W[(kb*32 + 16*(lane>>4) + e)*H
//                                               + ntile*16 + (lane&15)]
// so each lane's 32x16 B fragment is one contiguous 32-byte record.
// ---------------------------------------------------------------------------
__global__ __launch_bounds__(256) void prep_w_bf16(const float* __restrict__ W,
                                                   __bf16* __restrict__ Wb) {
    const int i     = blockIdx.x * 256 + threadIdx.x;   // [0, H*H)
    const int e     = i & 15;
    const int lane  = (i >> 4) & 31;
    const int ntile = (i >> 9) & 63;
    const int kb    = i >> 15;
    const int kk    = kb * 32 + 16 * (lane >> 4) + e;
    const int n     = ntile * 16 + (lane & 15);
    Wb[i] = (__bf16)W[(size_t)kk * HDIM + n];
}

// ---------------------------------------------------------------------------
// Fused: h = A*W + b ; h += residual ; LayerNorm(h) ; segment-sum into segSums
// grid = T/MB blocks of 256 threads (8 waves).
// ---------------------------------------------------------------------------
__global__ __launch_bounds__(256) void fused_gemm_ln_seg(
    const float*  __restrict__ A,       // [T,H]
    const float*  __restrict__ R,       // [T,H] residual
    const __bf16* __restrict__ Wb,      // pre-swizzled bf16 W fragments
    const float*  __restrict__ bD,      // [H]
    const float*  __restrict__ gamma,   // [H]
    const float*  __restrict__ beta,    // [H]
    const int*    __restrict__ segIds,  // [T] sorted
    float*        __restrict__ segSums) // [B,H] pre-zeroed
{
    __shared__ float sSeg[SEGCAP][HDIM];   // 32 KB segment accumulators
    __shared__ float sSum[NWAVE][MG];      // partial row sums
    __shared__ float sSsq[NWAVE][MG];      // partial row sum-of-squares

    const int tid   = threadIdx.x;
    const int wave  = tid >> 5;
    const int lane  = tid & 31;
    const int lhalf = lane >> 4;           // 0: lanes 0-15, 1: lanes 16-31
    const int l16   = lane & 15;
    const int row0  = blockIdx.x * MB;
    const int nBase = wave * NW;

    const v16bf* WbT = (const v16bf*)Wb;   // one v16bf per (kb, ntile, lane)

    const int firstSeg = segIds[row0];
    const int lastSeg  = segIds[row0 + MB - 1];
    const int nsegLds  = min(lastSeg - firstSeg + 1, SEGCAP);

    // zero the LDS segment accumulators we will use
    for (int i = tid; i < nsegLds * HDIM; i += 256)
        (&sSeg[0][0])[i] = 0.0f;
    __syncthreads();

    // per-lane column-constant preloads (cols owned by this lane)
    float gv[8], bev[8], bdv[8];
#pragma unroll
    for (int nt = 0; nt < 8; ++nt) {
        const int col = nBase + nt * 16 + l16;
        gv[nt]  = gamma[col];
        bev[nt] = beta[col];
        bdv[nt] = bD[col];
    }

    for (int mg = 0; mg < MB / MG; ++mg) {
        const int rBase = row0 + mg * MG;

        v8f acc[2][8];
        const v8f vzero = {};
#pragma unroll
        for (int t = 0; t < 2; ++t)
#pragma unroll
            for (int nt = 0; nt < 8; ++nt) acc[t][nt] = vzero;

        // ---- K loop: bf16 WMMA, fp32 accumulate --------------------------
        for (int k = 0; k < HDIM; k += 32) {
            const size_t kbBase = ((size_t)(k >> 5) * 64 + (nBase >> 4)) * 32 + lane;

            // A fragments (16x32 bf16 layout): one contiguous 16-float run
            v16bf afrag[2];
#pragma unroll
            for (int t = 0; t < 2; ++t) {
                const float* arow =
                    A + (size_t)(rBase + t * 16 + l16) * HDIM + k + 16 * lhalf;
                const float4 a0 = *(const float4*)(arow);
                const float4 a1 = *(const float4*)(arow + 4);
                const float4 a2 = *(const float4*)(arow + 8);
                const float4 a3 = *(const float4*)(arow + 12);
                afrag[t][0]  = (__bf16)a0.x;  afrag[t][1]  = (__bf16)a0.y;
                afrag[t][2]  = (__bf16)a0.z;  afrag[t][3]  = (__bf16)a0.w;
                afrag[t][4]  = (__bf16)a1.x;  afrag[t][5]  = (__bf16)a1.y;
                afrag[t][6]  = (__bf16)a1.z;  afrag[t][7]  = (__bf16)a1.w;
                afrag[t][8]  = (__bf16)a2.x;  afrag[t][9]  = (__bf16)a2.y;
                afrag[t][10] = (__bf16)a2.z;  afrag[t][11] = (__bf16)a2.w;
                afrag[t][12] = (__bf16)a3.x;  afrag[t][13] = (__bf16)a3.y;
                afrag[t][14] = (__bf16)a3.z;  afrag[t][15] = (__bf16)a3.w;
            }
            // B fragments: one 32-byte vector load each (pre-swizzled)
#pragma unroll
            for (int nt = 0; nt < 8; ++nt) {
                const v16bf bfrag = WbT[kbBase + (size_t)nt * 32];
#pragma unroll
                for (int t = 0; t < 2; ++t)
                    acc[t][nt] = __builtin_amdgcn_wmma_f32_16x16x32_bf16(
                        false, afrag[t], false, bfrag, (short)0, acc[t][nt],
                        false, false);
            }
        }

        // ---- epilogue: bias + residual, row stats ------------------------
        // C/D layout: VGPR v, lanes 0-15 -> row v, lanes 16-31 -> row v+8
#pragma unroll
        for (int t = 0; t < 2; ++t) {
#pragma unroll
            for (int v = 0; v < 8; ++v) {
                const int lr  = 16 * t + 8 * lhalf + v;     // local row in [0,32)
                const int row = rBase + lr;
                const float* rrow = R + (size_t)row * HDIM;
                float s = 0.f, sq = 0.f;
#pragma unroll
                for (int nt = 0; nt < 8; ++nt) {
                    float h = acc[t][nt][v] + bdv[nt] + rrow[nBase + nt * 16 + l16];
                    acc[t][nt][v] = h;
                    s += h; sq += h * h;
                }
                // reduce over the 16 lanes holding this row's column slice
#pragma unroll
                for (int off = 1; off < 16; off <<= 1) {
                    s  += __shfl_xor(s,  off, 32);
                    sq += __shfl_xor(sq, off, 32);
                }
                if (l16 == 0) {
                    sSum[wave][lr] = s;
                    sSsq[wave][lr] = sq;
                }
            }
        }
        __syncthreads();

        // ---- normalize + segment accumulate ------------------------------
#pragma unroll
        for (int t = 0; t < 2; ++t) {
#pragma unroll
            for (int v = 0; v < 8; ++v) {
                const int lr = 16 * t + 8 * lhalf + v;
                float s = 0.f, sq = 0.f;
#pragma unroll
                for (int w2 = 0; w2 < NWAVE; ++w2) { s += sSum[w2][lr]; sq += sSsq[w2][lr]; }
                const float mean = s * (1.0f / HDIM);
                const float var  = sq * (1.0f / HDIM) - mean * mean;
                const float rstd = rsqrtf(var + LN_EPS);

                const int row = rBase + lr;
                const int seg = segIds[row];
                const int sl  = seg - firstSeg;
#pragma unroll
                for (int nt = 0; nt < 8; ++nt) {
                    const int col = nBase + nt * 16 + l16;
                    const float y = (acc[t][nt][v] - mean) * rstd * gv[nt] + bev[nt];
                    if (sl < SEGCAP)
                        atomic_add_f32_lds(&sSeg[sl][col], y);
                    else
                        atomic_add_f32_global(&segSums[(size_t)seg * HDIM + col], y);
                }
            }
        }
        __syncthreads();   // protect sSum/sSsq reuse next m-group
    }

    // ---- flush LDS segment accumulators to global ------------------------
    for (int i = tid; i < nsegLds * HDIM; i += 256) {
        const int sl  = i >> 10;          // / HDIM
        const int col = i & (HDIM - 1);
        atomic_add_f32_global(&segSums[(size_t)(firstSeg + sl) * HDIM + col],
                              sSeg[sl][col]);
    }
}

// ---------------------------------------------------------------------------
__global__ void zero_ws(float* __restrict__ p, int n) {
    const int i = blockIdx.x * 256 + threadIdx.x;
    if (i < n) p[i] = 0.0f;
}

// segment token counts via LDS histogram; grid = T/1024, block = 256
__global__ void seg_counts(const int* __restrict__ ids, float* __restrict__ counts) {
    __shared__ float lc[B_SEG];
    const int tid = threadIdx.x;
    if (tid < B_SEG) lc[tid] = 0.0f;
    __syncthreads();
    const int base = blockIdx.x * 1024;
    for (int i = tid; i < 1024; i += 256)
        atomic_add_f32_lds(&lc[ids[base + i]], 1.0f);
    __syncthreads();
    if (tid < B_SEG && lc[tid] != 0.0f)
        atomic_add_f32_global(&counts[tid], lc[tid]);
}

// out[b,l] = (segSums[b,:]/counts[b]) . Wout[:,l] + bout[l]
__global__ void final_proj(const float* __restrict__ segSums,
                           const float* __restrict__ counts,
                           const float* __restrict__ Wout,   // [H,L]
                           const float* __restrict__ bout,   // [L]
                           float* __restrict__ out) {        // [B,L]
    const int t = threadIdx.x;
    if (t >= B_SEG * L_OUT) return;
    const int b = t >> 1, l = t & 1;
    const float inv = 1.0f / counts[b];
    const float* srow = segSums + (size_t)b * HDIM;
    float accv = 0.0f;
    for (int h = 0; h < HDIM; ++h)
        accv += srow[h] * Wout[h * L_OUT + l];
    out[t] = accv * inv + bout[l];
}

// ---------------------------------------------------------------------------
extern "C" void kernel_launch(void* const* d_in, const int* in_sizes, int n_in,
                              void* d_out, int out_size, void* d_ws, size_t ws_size,
                              hipStream_t stream) {
    const float* A   = (const float*)d_in[0];  // sequence_attention_embeddings
    const float* R   = (const float*)d_in[1];  // sequence_embeddings
    const float* Wd  = (const float*)d_in[2];  // W_dense
    const float* bD  = (const float*)d_in[3];  // b_dense
    const float* g   = (const float*)d_in[4];  // gamma
    const float* be  = (const float*)d_in[5];  // beta
    const float* Wo  = (const float*)d_in[6];  // W_out
    const float* bo  = (const float*)d_in[7];  // b_out
    const int*   ids = (const int*)  d_in[8];  // segment_ids
    float* out = (float*)d_out;

    // workspace layout: [Wb: H*H bf16 (2MB)] [segSums: B*H f32] [counts: B f32]
    __bf16* Wb     = (__bf16*)d_ws;
    float* segSums = (float*)((char*)d_ws + (size_t)HDIM * HDIM * sizeof(__bf16));
    float* counts  = segSums + B_SEG * HDIM;

    const int nz = B_SEG * HDIM + B_SEG;
    zero_ws<<<(nz + 255) / 256, 256, 0, stream>>>(segSums, nz);
    seg_counts<<<T_TOK / 1024, 256, 0, stream>>>(ids, counts);
    prep_w_bf16<<<(HDIM * HDIM) / 256, 256, 0, stream>>>(Wd, Wb);
    fused_gemm_ln_seg<<<T_TOK / MB, 256, 0, stream>>>(A, R, Wb, bD, g, be, ids, segSums);
    final_proj<<<1, 128, 0, stream>>>(segSums, counts, Wo, bo, out);
}
